// DNC_21930103013814
// MI455X (gfx1250) — matrix operand order repaired and checked by reading, not compiled, not run on hardware
//
#include <hip/hip_runtime.h>
#include <hip/hip_bf16.h>
#include <math.h>

// ---------------- problem constants ----------------
#define B_        32
#define T_        48
#define DIN       256
#define H_        512
#define G4H       2048        // 4*H
#define M_        256
#define W_        64
#define R_        4
#define IFACE_    471         // R*W + 3*W + 5*R + 3
#define IFACE_PAD 512         // padded so every GEMM N is a multiple of 128
#define EPS_      1e-6f

typedef _Float16 half_t;
typedef __attribute__((ext_vector_type(16))) _Float16 v16h;
typedef __attribute__((ext_vector_type(8)))  float    v8f;
typedef int v4i_ __attribute__((ext_vector_type(4)));
// exact parameter types for the gfx1250 async global->LDS builtin:
//   (int4 AS1*, int4 AS3*, imm offset, imm cpol)
typedef __attribute__((address_space(1))) v4i_* as1_v4i_p;
typedef __attribute__((address_space(3))) v4i_* as3_v4i_p;

// optional gfx1250 async global->LDS path (ASYNCcnt-tracked), guarded so a
// toolchain without the builtins falls back to a plain vector copy.
#ifndef HAVE_ASYNC_LDS
#if defined(__has_builtin)
#if __has_builtin(__builtin_amdgcn_global_load_async_to_lds_b128) && \
    __has_builtin(__builtin_amdgcn_s_wait_asynccnt)
#define HAVE_ASYNC_LDS 1
#else
#define HAVE_ASYNC_LDS 0
#endif
#else
#define HAVE_ASYNC_LDS 0
#endif
#endif

__device__ __forceinline__ float sigmoidf_(float x) { return 1.0f / (1.0f + expf(-x)); }
__device__ __forceinline__ float softplusf_(float x) { return x > 20.0f ? x : log1pf(expf(x)); }

// ---------------- prep kernels ----------------
__global__ void cvt_f16_pad_kernel(const float* __restrict__ src, half_t* __restrict__ dst,
                                   int n_src, int n_dst) {
    int i = blockIdx.x * blockDim.x + threadIdx.x;
    if (i < n_dst) dst[i] = (i < n_src) ? (half_t)src[i] : (half_t)0.0f;
}

__global__ void zero_f32_kernel(float* __restrict__ p, int n) {
    int i = blockIdx.x * blockDim.x + threadIdx.x;
    if (i < n) p[i] = 0.0f;
}

__global__ void zero_f16_kernel(half_t* __restrict__ p, int n) {
    int i = blockIdx.x * blockDim.x + threadIdx.x;
    if (i < n) p[i] = (half_t)0.0f;
}

// ---------------- WMMA GEMM ----------------
// Y[b][n] (+)= sum_k X[b][k] * Wt[n][k] (+ bias[n]); optional f16 mirror Yh.
// X is f16 [B][ldx]. Block = 128 threads (4 waves); each wave owns 2 N-tiles
// sharing one A fragment; the 16xK A tile is staged once per block in LDS.
// grid.x = N/128, grid.y = B/16 (=2).  Fragment layouts per ISA 7.12.2.
__global__ void gemm_wmma_f16(const half_t* __restrict__ X, int ldx,
                              const half_t* __restrict__ Wt, int ldw,
                              const float* __restrict__ bias, int bias_n,
                              float* __restrict__ Y, int ldy,
                              half_t* __restrict__ Yh,
                              int K, int accumulate) {
    __shared__ __align__(16) half_t As[16 * 512];   // K <= 512
    const int tid  = threadIdx.x;
    const int wave = tid >> 5;
    const int lane = tid & 31;
    const int hf   = lane >> 4;                 // lane half
    const int l    = lane & 15;
    const int tb   = blockIdx.y * 16;           // batch-tile base (M dim)
    const int tn0  = blockIdx.x * 128 + wave * 32;  // this wave's first N-tile

    // ---- stage X tile (16 x K halfs) into LDS ----
    const int nvec = (16 * K) >> 3;             // 16-byte vectors
    const int kv8  = K >> 3;
#if HAVE_ASYNC_LDS
    for (int idx = tid; idx < nvec; idx += 128) {
        int m = idx / kv8, kv = idx - m * kv8;
        const half_t* src = X + (size_t)(tb + m) * ldx + kv * 8;
        half_t*       dst = As + m * K + kv * 8;
        __builtin_amdgcn_global_load_async_to_lds_b128((as1_v4i_p)src, (as3_v4i_p)dst, 0, 0);
    }
    __builtin_amdgcn_s_wait_asynccnt(0);
    __syncthreads();
#else
    for (int idx = tid; idx < nvec; idx += 128) {
        int m = idx / kv8, kv = idx - m * kv8;
        *(uint4*)(As + m * K + kv * 8) =
            *(const uint4*)(X + (size_t)(tb + m) * ldx + kv * 8);
    }
    __syncthreads();
#endif

    // ---- accumulator init (bias / accumulate) ----
    v8f acc0, acc1;
    const int n0 = tn0 + l;
    const int n1 = tn0 + 16 + l;
#pragma unroll
    for (int r = 0; r < 8; ++r) {
        int brow = tb + r + hf * 8;
        float c0 = (bias && n0 < bias_n) ? bias[n0] : 0.0f;
        float c1 = (bias && n1 < bias_n) ? bias[n1] : 0.0f;
        if (accumulate) {
            c0 += Y[(size_t)brow * ldy + n0];
            c1 += Y[(size_t)brow * ldy + n1];
        }
        acc0[r] = c0;
        acc1[r] = c1;
    }

    const half_t* wrow0 = Wt + (size_t)n0 * ldw;
    const half_t* wrow1 = Wt + (size_t)n1 * ldw;
    __builtin_prefetch(wrow0, 0, 1);            // global_prefetch_b8
    __builtin_prefetch(wrow1, 0, 1);

    for (int k0 = 0; k0 < K; k0 += 32) {
        v16h a, b0, b1;
#pragma unroll
        for (int i = 0; i < 16; ++i) {
            // A 16x32 f16: elems 0-7 -> K=k0+0..7 (+8 for lanes 16-31),
            //              elems 8-15 -> K=k0+16..23 (+8 for lanes 16-31)
            int ka = k0 + ((i < 8) ? i : i + 8) + hf * 8;
            a[i] = As[l * K + ka];
            // B 32x16 f16: lane = N; same K order as A (elems 0-7 -> k0+0..7, 8-15 -> k0+16..23, +8 for lanes 16-31)
            int kb = ka;
            b0[i] = wrow0[kb];
            b1[i] = wrow1[kb];
        }
        acc0 = __builtin_amdgcn_wmma_f32_16x16x32_f16(false, a, false, b0, (short)0, acc0,
                                                      false, false);
        acc1 = __builtin_amdgcn_wmma_f32_16x16x32_f16(false, a, false, b1, (short)0, acc1,
                                                      false, false);
    }

#pragma unroll
    for (int r = 0; r < 8; ++r) {
        int brow = tb + r + hf * 8;
        Y[(size_t)brow * ldy + n0] = acc0[r];
        Y[(size_t)brow * ldy + n1] = acc1[r];
        if (Yh) {
            Yh[(size_t)brow * ldy + n0] = (half_t)acc0[r];
            Yh[(size_t)brow * ldy + n1] = (half_t)acc1[r];
        }
    }
}

// ---------------- LSTM pointwise (h emitted as f16 for the next GEMMs) ----------------
__global__ void lstm_pointwise_kernel(const float* __restrict__ g,
                                      float* __restrict__ c_state,
                                      half_t* __restrict__ h_state) {
    int i = blockIdx.x * blockDim.x + threadIdx.x;
    if (i >= B_ * H_) return;
    int b = i / H_, n = i - b * H_;
    const float* gb = g + (size_t)b * G4H;
    float gi = gb[n], gf = gb[H_ + n], gc = gb[2 * H_ + n], go = gb[3 * H_ + n];
    float c2 = sigmoidf_(gf) * c_state[i] + sigmoidf_(gi) * tanhf(gc);
    c_state[i] = c2;
    h_state[i] = (half_t)(sigmoidf_(go) * tanhf(c2));
}

// ---------------- memory module ----------------
// interface layout in v[b][IFACE_PAD]:
//   rk: [0,256) (R,W)  rs: [256,260)  wk: [260,324)  ws: 324
//   er: [325,389)  wv: [389,453)  fg: [453,457)  ag: 457  wg: 458  rm: [459,471)

__global__ void usage_kernel(const float* __restrict__ v, const float* __restrict__ ww_prev,
                             const float* __restrict__ rw_prev, float* __restrict__ usage) {
    int i = blockIdx.x * blockDim.x + threadIdx.x;
    if (i >= B_ * M_) return;
    int b = i / M_, m = i - b * M_;
    const float* vb = v + (size_t)b * IFACE_PAD;
    float u = usage[i];
    u = u + (1.0f - u) * ww_prev[i];
    float psi = 1.0f;
#pragma unroll
    for (int r = 0; r < R_; ++r) {
        float fg = sigmoidf_(vb[453 + r]);
        psi *= (1.0f - fg * rw_prev[((size_t)b * R_ + r) * M_ + m]);
    }
    usage[i] = u * psi;
}

__global__ void content_write_kernel(const float* __restrict__ v,
                                     const float* __restrict__ mem,
                                     float* __restrict__ cw_w) {
    __shared__ float red[M_];
    __shared__ float kvec[W_];
    __shared__ float knorm_s;
    int b = blockIdx.x, m = threadIdx.x;
    const float* vb = v + (size_t)b * IFACE_PAD;
    if (m < W_) kvec[m] = tanhf(vb[260 + m]);
    __syncthreads();
    if (m == 0) {
        float s = 0.0f;
        for (int w = 0; w < W_; ++w) s += kvec[w] * kvec[w];
        knorm_s = sqrtf(s) + EPS_;
    }
    __syncthreads();
    const float* mrow = mem + ((size_t)b * M_ + m) * W_;
    float dot = 0.0f, nn = 0.0f;
    for (int w = 0; w < W_; ++w) { float mv = mrow[w]; dot += mv * kvec[w]; nn += mv * mv; }
    float beta  = softplusf_(vb[324]);
    float score = dot / ((sqrtf(nn) + EPS_) * knorm_s) * beta;
    red[m] = score; __syncthreads();
    for (int s = 128; s > 0; s >>= 1) { if (m < s) red[m] = fmaxf(red[m], red[m + s]); __syncthreads(); }
    float mx = red[0]; __syncthreads();
    float e = expf(score - mx);
    red[m] = e; __syncthreads();
    for (int s = 128; s > 0; s >>= 1) { if (m < s) red[m] += red[m + s]; __syncthreads(); }
    cw_w[(size_t)b * M_ + m] = e / red[0];
}

// allocation weights (stable argsort via (value,index) bitonic sort) + write weights
__global__ void alloc_ww_kernel(const float* __restrict__ v, const float* __restrict__ usage,
                                const float* __restrict__ cw_w, float* __restrict__ ww,
                                float* __restrict__ wwsum) {
    __shared__ float sval[M_];
    __shared__ int   sidx[M_];
    __shared__ float cp[M_];
    __shared__ float allocw[M_];
    __shared__ float redsum[M_];
    int b = blockIdx.x, t = threadIdx.x;
    float u = EPS_ + (1.0f - EPS_) * usage[(size_t)b * M_ + t];
    sval[t] = u; sidx[t] = t;
    __syncthreads();
    for (int k = 2; k <= M_; k <<= 1) {
        for (int j = k >> 1; j > 0; j >>= 1) {
            int ixj = t ^ j;
            if (ixj > t) {
                bool up = ((t & k) == 0);
                float v0 = sval[t], v1 = sval[ixj];
                int   i0 = sidx[t], i1 = sidx[ixj];
                bool gt = (v0 > v1) || (v0 == v1 && i0 > i1);
                bool sw = up ? gt : !gt;
                if (sw) { sval[t] = v1; sval[ixj] = v0; sidx[t] = i1; sidx[ixj] = i0; }
            }
            __syncthreads();
        }
    }
    if (t == 0) {
        float p = 1.0f;
        for (int j = 0; j < M_; ++j) { cp[j] = p; p *= sval[j]; }
    }
    __syncthreads();
    allocw[sidx[t]] = (1.0f - sval[t]) * cp[t];
    __syncthreads();
    const float* vb = v + (size_t)b * IFACE_PAD;
    float ag = sigmoidf_(vb[457]);
    float wg = sigmoidf_(vb[458]);
    float w  = wg * (ag * allocw[t] + (1.0f - ag) * cw_w[(size_t)b * M_ + t]);
    ww[(size_t)b * M_ + t] = w;
    redsum[t] = w; __syncthreads();
    for (int s = 128; s > 0; s >>= 1) { if (t < s) redsum[t] += redsum[t + s]; __syncthreads(); }
    if (t == 0) wwsum[b] = redsum[0];
}

__global__ void mem_write_kernel(const float* __restrict__ v, const float* __restrict__ ww,
                                 float* __restrict__ mem) {
    int i = blockIdx.x * blockDim.x + threadIdx.x;
    if (i >= B_ * M_ * W_) return;
    int b = i / (M_ * W_);
    int rem = i - b * (M_ * W_);
    int m = rem / W_, w = rem - m * W_;
    const float* vb = v + (size_t)b * IFACE_PAD;
    float er  = sigmoidf_(vb[325 + w]);
    float wv  = tanhf(vb[389 + w]);
    float wwv = ww[(size_t)b * M_ + m];
    mem[i] = mem[i] * (1.0f - wwv * er) + wwv * wv;
}

__global__ void link_kernel(const float* __restrict__ ww, const float* __restrict__ prec,
                            float* __restrict__ link) {
    size_t i = (size_t)blockIdx.x * blockDim.x + threadIdx.x;
    if (i >= (size_t)B_ * M_ * M_) return;
    int b = (int)(i / (M_ * M_));
    int rem = (int)(i - (size_t)b * M_ * M_);
    int ii = rem / M_, jj = rem - ii * M_;
    if (ii == jj) { link[i] = 0.0f; return; }
    float wi = ww[(size_t)b * M_ + ii], wj = ww[(size_t)b * M_ + jj];
    link[i] = (1.0f - wi - wj) * link[i] + wi * prec[(size_t)b * M_ + jj];
}

__global__ void prec_kernel(const float* __restrict__ ww, const float* __restrict__ wwsum,
                            float* __restrict__ prec) {
    int i = blockIdx.x * blockDim.x + threadIdx.x;
    if (i >= B_ * M_) return;
    int b = i / M_;
    prec[i] = (1.0f - wwsum[b]) * prec[i] + ww[i];
}

// one block per (b,r); rw_prev staged in LDS; bwd pass is coalesced over threads
__global__ void fwdbwd_kernel(const float* __restrict__ link, const float* __restrict__ rw_prev,
                              float* __restrict__ fwd, float* __restrict__ bwd) {
    __shared__ float rwp[M_];
    int b = blockIdx.x / R_, r = blockIdx.x - b * R_;
    int i = threadIdx.x;                     // 0..255
    rwp[i] = rw_prev[((size_t)b * R_ + r) * M_ + i];
    __syncthreads();
    const float* L = link + (size_t)b * M_ * M_;
    float f = 0.0f, bb = 0.0f;
    for (int j = 0; j < M_; ++j) {
        f  += L[(size_t)i * M_ + j] * rwp[j];   // row i, contiguous per thread
        bb += rwp[j] * L[(size_t)j * M_ + i];   // column i, coalesced across threads
    }
    size_t o = ((size_t)b * R_ + r) * M_ + i;
    fwd[o] = f; bwd[o] = bb;
}

__global__ void content_read_kernel(const float* __restrict__ v, const float* __restrict__ mem,
                                    float* __restrict__ cw_r) {
    __shared__ float red[M_];
    __shared__ float kvec[W_];
    __shared__ float knorm_s;
    int b = blockIdx.x / R_, r = blockIdx.x - b * R_;
    int m = threadIdx.x;
    const float* vb = v + (size_t)b * IFACE_PAD;
    if (m < W_) kvec[m] = tanhf(vb[r * W_ + m]);
    __syncthreads();
    if (m == 0) {
        float s = 0.0f;
        for (int w = 0; w < W_; ++w) s += kvec[w] * kvec[w];
        knorm_s = sqrtf(s) + EPS_;
    }
    __syncthreads();
    const float* mrow = mem + ((size_t)b * M_ + m) * W_;
    float dot = 0.0f, nn = 0.0f;
    for (int w = 0; w < W_; ++w) { float mv = mrow[w]; dot += mv * kvec[w]; nn += mv * mv; }
    float beta  = softplusf_(vb[256 + r]);
    float score = dot / ((sqrtf(nn) + EPS_) * knorm_s) * beta;
    red[m] = score; __syncthreads();
    for (int s = 128; s > 0; s >>= 1) { if (m < s) red[m] = fmaxf(red[m], red[m + s]); __syncthreads(); }
    float mx = red[0]; __syncthreads();
    float e = expf(score - mx);
    red[m] = e; __syncthreads();
    for (int s = 128; s > 0; s >>= 1) { if (m < s) red[m] += red[m + s]; __syncthreads(); }
    cw_r[((size_t)b * R_ + r) * M_ + m] = e / red[0];
}

__global__ void rw_kernel(const float* __restrict__ v, const float* __restrict__ fwd,
                          const float* __restrict__ bwd, const float* __restrict__ cw_r,
                          float* __restrict__ rw) {
    int i = blockIdx.x * blockDim.x + threadIdx.x;
    if (i >= B_ * R_ * M_) return;
    int b = i / (R_ * M_);
    int rem = i - b * (R_ * M_);
    int r = rem / M_;
    const float* vb = v + (size_t)b * IFACE_PAD + 459 + r * 3;
    float a0 = vb[0], a1 = vb[1], a2 = vb[2];
    float mx = fmaxf(a0, fmaxf(a1, a2));
    float e0 = expf(a0 - mx), e1 = expf(a1 - mx), e2 = expf(a2 - mx);
    float inv = 1.0f / (e0 + e1 + e2);
    rw[i] = (e0 * bwd[i] + e1 * fwd[i] + e2 * cw_r[i]) * inv;
}

// read vectors emitted directly as f16 (consumed only by the W_rv GEMM)
__global__ void rv_kernel(const float* __restrict__ rw, const float* __restrict__ mem,
                          half_t* __restrict__ rv16) {
    int i = blockIdx.x * blockDim.x + threadIdx.x;
    if (i >= B_ * R_ * W_) return;
    int b = i / (R_ * W_);
    int rem = i - b * (R_ * W_);
    int r = rem / W_, w = rem - r * W_;
    const float* rwb = rw + ((size_t)b * R_ + r) * M_;
    const float* mb  = mem + (size_t)b * M_ * W_ + w;
    float s = 0.0f;
    for (int m = 0; m < M_; ++m) s += rwb[m] * mb[(size_t)m * W_];
    rv16[i] = (half_t)s;
}

// ---------------- host orchestration ----------------
extern "C" void kernel_launch(void* const* d_in, const int* in_sizes, int n_in,
                              void* d_out, int out_size, void* d_ws, size_t ws_size,
                              hipStream_t stream) {
    const float* x        = (const float*)d_in[0];
    const float* W_ih0    = (const float*)d_in[1];
    const float* b_ih0    = (const float*)d_in[2];
    const float* W_hh0    = (const float*)d_in[3];
    const float* b_hh0    = (const float*)d_in[4];
    const float* W_ih1    = (const float*)d_in[5];
    const float* b_ih1    = (const float*)d_in[6];
    const float* W_hh1    = (const float*)d_in[7];
    const float* b_hh1    = (const float*)d_in[8];
    const float* W_iface  = (const float*)d_in[9];
    const float* b_iface  = (const float*)d_in[10];
    const float* W_memout = (const float*)d_in[11];
    const float* b_memout = (const float*)d_in[12];
    const float* W_rv     = (const float*)d_in[13];
    const float* b_rv     = (const float*)d_in[14];
    const float* W_out    = (const float*)d_in[15];
    const float* b_out    = (const float*)d_in[16];
    float* y_out          = (float*)d_out;

    // bump allocator over d_ws
    char* p = (char*)d_ws;
    auto align16 = [&]() { uintptr_t u = (uintptr_t)p; p = (char*)((u + 15) & ~(uintptr_t)15); };
    auto alloc_f = [&](size_t n) { align16(); float* r = (float*)p; p += n * sizeof(float); return r; };
    auto alloc_h = [&](size_t n) { align16(); half_t* r = (half_t*)p; p += n * sizeof(half_t); return r; };

    // zero-initialized f32 state block (contiguous)
    float* state0 = alloc_f(0);
    float* c0    = alloc_f((size_t)B_ * H_);
    float* c1    = alloc_f((size_t)B_ * H_);
    float* mem   = alloc_f((size_t)B_ * M_ * W_);
    float* link  = alloc_f((size_t)B_ * M_ * M_);
    float* prec  = alloc_f((size_t)B_ * M_);
    float* rw    = alloc_f((size_t)B_ * R_ * M_);
    float* ww    = alloc_f((size_t)B_ * M_);
    float* usage = alloc_f((size_t)B_ * M_);
    size_t n_state = (size_t)(((float*)p) - state0);

    // zero-initialized f16 state (h0, h1 contiguous)
    half_t* h0_16 = alloc_h((size_t)B_ * H_);
    half_t* h1_16 = alloc_h((size_t)B_ * H_);
    size_t n_hstate = (size_t)2 * B_ * H_;

    // per-step scratch
    float*  g      = alloc_f((size_t)B_ * G4H);
    float*  v      = alloc_f((size_t)B_ * IFACE_PAD);
    float*  outb   = alloc_f((size_t)B_ * H_);
    half_t* outb16 = alloc_h((size_t)B_ * H_);
    float*  cw_w   = alloc_f((size_t)B_ * M_);
    float*  cw_r   = alloc_f((size_t)B_ * R_ * M_);
    float*  fwd    = alloc_f((size_t)B_ * R_ * M_);
    float*  bwd    = alloc_f((size_t)B_ * R_ * M_);
    half_t* rv16   = alloc_h((size_t)B_ * R_ * W_);
    float*  wwsum  = alloc_f((size_t)B_);

    // f16 inputs / weights
    half_t* x16       = alloc_h((size_t)B_ * T_ * DIN);
    half_t* Wih0_h    = alloc_h((size_t)G4H * DIN);
    half_t* Whh0_h    = alloc_h((size_t)G4H * H_);
    half_t* Wih1_h    = alloc_h((size_t)G4H * H_);
    half_t* Whh1_h    = alloc_h((size_t)G4H * H_);
    half_t* Wiface_h  = alloc_h((size_t)IFACE_PAD * H_);
    half_t* Wmemout_h = alloc_h((size_t)H_ * H_);
    half_t* Wrv_h     = alloc_h((size_t)H_ * (R_ * W_));
    half_t* Wout_h    = alloc_h((size_t)DIN * H_);

    auto cvt = [&](const float* src, half_t* dst, int n_src, int n_dst) {
        cvt_f16_pad_kernel<<<(n_dst + 255) / 256, 256, 0, stream>>>(src, dst, n_src, n_dst);
    };
    cvt(x,        x16,       B_ * T_ * DIN,  B_ * T_ * DIN);
    cvt(W_ih0,    Wih0_h,    G4H * DIN,      G4H * DIN);
    cvt(W_hh0,    Whh0_h,    G4H * H_,       G4H * H_);
    cvt(W_ih1,    Wih1_h,    G4H * H_,       G4H * H_);
    cvt(W_hh1,    Whh1_h,    G4H * H_,       G4H * H_);
    cvt(W_iface,  Wiface_h,  IFACE_ * H_,    IFACE_PAD * H_);
    cvt(W_memout, Wmemout_h, H_ * H_,        H_ * H_);
    cvt(W_rv,     Wrv_h,     H_ * R_ * W_,   H_ * R_ * W_);
    cvt(W_out,    Wout_h,    DIN * H_,       DIN * H_);

    zero_f32_kernel<<<((int)n_state + 255) / 256, 256, 0, stream>>>(state0, (int)n_state);
    zero_f16_kernel<<<((int)n_hstate + 255) / 256, 256, 0, stream>>>(h0_16, (int)n_hstate);

    auto gemm = [&](const half_t* X, int ldx, const half_t* Wt, int ldw,
                    const float* bias, int bias_n, float* Y, int ldy, half_t* Yh,
                    int K, int N, int acc) {
        gemm_wmma_f16<<<dim3(N / 128, B_ / 16), 128, 0, stream>>>(X, ldx, Wt, ldw, bias, bias_n,
                                                                  Y, ldy, Yh, K, acc);
    };

    for (int t = 0; t < T_; ++t) {
        // ---- LSTM layer 0 ----
        gemm(x16 + (size_t)t * DIN, T_ * DIN, Wih0_h, DIN, b_ih0, G4H, g, G4H, nullptr,
             DIN, G4H, 0);
        gemm(h0_16, H_, Whh0_h, H_, b_hh0, G4H, g, G4H, nullptr, H_, G4H, 1);
        lstm_pointwise_kernel<<<(B_ * H_ + 255) / 256, 256, 0, stream>>>(g, c0, h0_16);
        // ---- LSTM layer 1 ----
        gemm(h0_16, H_, Wih1_h, H_, b_ih1, G4H, g, G4H, nullptr, H_, G4H, 0);
        gemm(h1_16, H_, Whh1_h, H_, b_hh1, G4H, g, G4H, nullptr, H_, G4H, 1);
        lstm_pointwise_kernel<<<(B_ * H_ + 255) / 256, 256, 0, stream>>>(g, c1, h1_16);
        // ---- interface + memout projections ----
        gemm(h1_16, H_, Wiface_h,  H_, b_iface,  IFACE_, v,    IFACE_PAD, nullptr,
             H_, IFACE_PAD, 0);
        gemm(h1_16, H_, Wmemout_h, H_, b_memout, H_,     outb, H_,        nullptr,
             H_, H_, 0);
        // ---- memory step ----
        usage_kernel<<<(B_ * M_ + 255) / 256, 256, 0, stream>>>(v, ww, rw, usage);
        content_write_kernel<<<B_, M_, 0, stream>>>(v, mem, cw_w);
        alloc_ww_kernel<<<B_, M_, 0, stream>>>(v, usage, cw_w, ww, wwsum);
        mem_write_kernel<<<(B_ * M_ * W_ + 255) / 256, 256, 0, stream>>>(v, ww, mem);
        link_kernel<<<(B_ * M_ * M_ + 255) / 256, 256, 0, stream>>>(ww, prec, link);
        prec_kernel<<<(B_ * M_ + 255) / 256, 256, 0, stream>>>(ww, wwsum, prec);
        fwdbwd_kernel<<<B_ * R_, M_, 0, stream>>>(link, rw, fwd, bwd);
        content_read_kernel<<<B_ * R_, M_, 0, stream>>>(v, mem, cw_r);
        rw_kernel<<<(B_ * R_ * M_ + 255) / 256, 256, 0, stream>>>(v, fwd, bwd, cw_r, rw);
        rv_kernel<<<(B_ * R_ * W_ + 255) / 256, 256, 0, stream>>>(rw, mem, rv16);
        // ---- output projections ----
        gemm(rv16, R_ * W_, Wrv_h, R_ * W_, b_rv, H_, outb, H_, outb16, R_ * W_, H_, 1);
        gemm(outb16, H_, Wout_h, H_, b_out, DIN, y_out + (size_t)t * DIN, T_ * DIN, nullptr,
             H_, DIN, 0);
    }
    (void)in_sizes; (void)n_in; (void)out_size; (void)ws_size;
}
